// HierarchicalReasoningModel_27015344292253
// MI455X (gfx1250) — matrix operand order repaired and checked by reading, not compiled, not run on hardware
//
#include <hip/hip_runtime.h>
#include <hip/hip_bf16.h>
#include <math.h>

// ---------------------------------------------------------------------------
// Model dims (match reference)
// ---------------------------------------------------------------------------
#define H_DIM 1024
#define D_DIM 512
#define O_DIM 256
#define B_DIM 2048
#define S_DIM 128
#define NUM_STEPS 3
#define L_ITERS 5

typedef __bf16 bf16_t;
typedef __attribute__((ext_vector_type(16))) __bf16 v16bf;
typedef __attribute__((ext_vector_type(8)))  __bf16 v8bf;
typedef __attribute__((ext_vector_type(4)))  __bf16 v4bf;
typedef __attribute__((ext_vector_type(8)))  float  v8f;
typedef int i32x4 __attribute__((vector_size(16)));

// ---------------------------------------------------------------------------
// CDNA5 async global->LDS copy (ASYNCcnt path), with synchronous fallback.
// Builtin signature (from toolchain diagnostic):
//   void __builtin_amdgcn_global_load_async_to_lds_b128(
//       i32x4 AS(1)* src_global, i32x4 AS(3)* dst_lds, imm offset, imm cpol)
// ---------------------------------------------------------------------------
#if defined(__has_builtin)
#if __has_builtin(__builtin_amdgcn_global_load_async_to_lds_b128)
#define HAVE_ASYNC_LDS 1
#endif
#endif

__device__ __forceinline__ void stage_chunk16(bf16_t* s, const bf16_t* g) {
#if defined(HAVE_ASYNC_LDS)
  __builtin_amdgcn_global_load_async_to_lds_b128(
      (__attribute__((address_space(1))) i32x4*)g,
      (__attribute__((address_space(3))) i32x4*)s, 0, 0);
#else
  *(uint4*)s = *(const uint4*)g;
#endif
}

__device__ __forceinline__ void wait_stage() {
#if defined(HAVE_ASYNC_LDS)
#if __has_builtin(__builtin_amdgcn_s_wait_asynccnt)
  __builtin_amdgcn_s_wait_asynccnt(0);
#else
  asm volatile("s_wait_asynccnt 0" ::: "memory");
#endif
#endif
}

// ---------------------------------------------------------------------------
// GEMM: C[M,N] = act(A[M,K] @ W[N,K]^T + bias)
//   A : bf16 activations (row stride lda elements)
//   W : bf16 weights, [N,K] row-major (K contiguous == B-operand column)
// Block tile 128(M) x 128(N), K-step 32, double-buffered LDS staging.
// 8 waves = 4(M) x 2(N); wave tile 32x64 -> 2 A frags x 4 B frags, 8 wmma.
// Requirements: M%128==0, N%128==0, K%32==0 (true for all GEMMs here).
// ---------------------------------------------------------------------------
#define KSTEP 32
#define BM 128
#define BN 128
#define LDSPITCH 40  // 32 + 8 pad, bf16 elements per staged row

template <bool RELU, bool OUTBF16>
__global__ __launch_bounds__(256) void gemm_bf16_wmma(
    const bf16_t* __restrict__ A, long lda,
    const bf16_t* __restrict__ W,
    const float* __restrict__ bias,
    void* __restrict__ Cout, int M, int N, int K) {
  __shared__ alignas(16) bf16_t sA[2][BM * LDSPITCH];
  __shared__ alignas(16) bf16_t sB[2][BN * LDSPITCH];

  const int tid   = threadIdx.x;
  const int lane  = tid & 31;
  const int wave  = tid >> 5;
  const int waveM = wave >> 1;  // 0..3
  const int waveN = wave & 1;   // 0..1
  const int half  = lane >> 4;  // 0|1
  const int r16   = lane & 15;

  const int rowStart = blockIdx.y * BM;
  const int colStart = blockIdx.x * BN;

  const bf16_t* gA = A + (long)rowStart * lda;
  const bf16_t* gB = W + (long)colStart * K;

  // staging: 512 16B-chunks per operand tile -> 2 A + 2 B chunks per thread
  // chunk c: row = c>>2, K-segment = (c&3)*8 elements
  auto stage = [&](bf16_t* sAb, bf16_t* sBb, int kofs) {
#pragma unroll
    for (int u = 0; u < 2; ++u) {
      const int c  = tid * 2 + u;
      const int r  = c >> 2;
      const int s8 = (c & 3) * 8;
      stage_chunk16(sAb + r * LDSPITCH + s8, gA + (long)r * lda + kofs + s8);
      stage_chunk16(sBb + r * LDSPITCH + s8, gB + (long)r * K + kofs + s8);
    }
  };

  // preload k=0
  stage(sA[0], sB[0], 0);
  wait_stage();
  __syncthreads();

  v8f acc[2][4] = {};

  const int kIters = K / KSTEP;
  for (int ki = 0; ki < kIters; ++ki) {
    const int cur = ki & 1;
    const int nxt = cur ^ 1;
    const bool more = (ki + 1) < kIters;
    if (more) stage(sA[nxt], sB[nxt], (ki + 1) * KSTEP);

    // ---- A fragments (ISA 16-bit A layout):
    //   lanes 0-15 : elems 0..7 = K0..7,  elems 8..15 = K16..23
    //   lanes16-31 : elems 0..7 = K8..15, elems 8..15 = K24..31
    v16bf afrag[2];
#pragma unroll
    for (int m = 0; m < 2; ++m) {
      const int row = waveM * 32 + m * 16 + r16;
      const bf16_t* pa = &sA[cur][row * LDSPITCH + half * 8];
      const v8bf lo = *(const v8bf*)pa;
      const v8bf hi = *(const v8bf*)(pa + 16);
      afrag[m] = __builtin_shufflevector(lo, hi, 0, 1, 2, 3, 4, 5, 6, 7, 8, 9,
                                         10, 11, 12, 13, 14, 15);
    }
    // ---- B fragments: lane holds column, same K order as A (K0..7/K16..23 | K8..15/K24..31)
    v16bf bfrag[4];
#pragma unroll
    for (int s = 0; s < 4; ++s) {
      const int col = waveN * 64 + s * 16 + r16;
      const bf16_t* pb = &sB[cur][col * LDSPITCH + half * 8];
      const v8bf lo = *(const v8bf*)pb;
      const v8bf hi = *(const v8bf*)(pb + 16);
      bfrag[s] = __builtin_shufflevector(lo, hi, 0, 1, 2, 3, 4, 5, 6, 7, 8, 9,
                                         10, 11, 12, 13, 14, 15);
    }
#pragma unroll
    for (int m = 0; m < 2; ++m)
#pragma unroll
      for (int s = 0; s < 4; ++s)
        acc[m][s] = __builtin_amdgcn_wmma_f32_16x16x32_bf16(
            false, afrag[m], false, bfrag[s], (short)0, acc[m][s], false, false);

    if (more) {
      wait_stage();
      __syncthreads();
    }
  }

  // ---- epilogue: C layout: lane r16 = column, VGPR i -> M = i (+8 if half)
  const int mb = rowStart + waveM * 32 + (half ? 8 : 0);
#pragma unroll
  for (int m = 0; m < 2; ++m) {
#pragma unroll
    for (int s = 0; s < 4; ++s) {
      const int   col = colStart + waveN * 64 + s * 16 + r16;
      const float bv  = bias ? bias[col] : 0.0f;
#pragma unroll
      for (int i = 0; i < 8; ++i) {
        float v = acc[m][s][i] + bv;
        if (RELU) v = v > 0.0f ? v : 0.0f;
        const long idx = (long)(mb + m * 16 + i) * N + col;
        if (OUTBF16)
          ((bf16_t*)Cout)[idx] = (bf16_t)v;
        else
          ((float*)Cout)[idx] = v;
      }
    }
  }
}

// ---------------------------------------------------------------------------
// Fused GRU gates: h = (1-z)*tanh(in + r*hn) + z*h   (gate order r, z, n)
// Writes f32 h (carried state) and bf16 copy (next GEMM's A operand).
// ---------------------------------------------------------------------------
__global__ __launch_bounds__(256) void gru_fuse_kernel(
    const float* __restrict__ gi, const float* __restrict__ gh,
    float* __restrict__ h, bf16_t* __restrict__ hbf, int n) {
  const int i = blockIdx.x * 256 + threadIdx.x;
  if (i >= n) return;
  const int  b    = i >> 10;  // / H_DIM
  const int  j    = i & (H_DIM - 1);
  const long base = (long)b * (3 * H_DIM) + j;

  const float ir  = gi[base],              hr = gh[base];
  const float iz  = gi[base + H_DIM],      hz = gh[base + H_DIM];
  const float in_ = gi[base + 2 * H_DIM],  hn = gh[base + 2 * H_DIM];

  const float r  = 1.0f / (1.0f + __expf(-(ir + hr)));
  const float z  = 1.0f / (1.0f + __expf(-(iz + hz)));
  const float nn = tanhf(in_ + r * hn);
  const float hv = (1.0f - z) * nn + z * h[i];
  h[i]   = hv;
  hbf[i] = (bf16_t)hv;
}

// ---------------------------------------------------------------------------
// LayerNorm over H=1024: one block of 256 threads per row (float4/thread).
// Input f32, output bf16 (consumed only as GEMM A operand).
// ---------------------------------------------------------------------------
__global__ __launch_bounds__(256) void layernorm_kernel(
    const float* __restrict__ x, const float* __restrict__ g,
    const float* __restrict__ bb, bf16_t* __restrict__ y) {
  __shared__ float red[256];
  const int t = threadIdx.x;
  const float4 v = ((const float4*)(x + (size_t)blockIdx.x * H_DIM))[t];

  red[t] = v.x + v.y + v.z + v.w;
  __syncthreads();
#pragma unroll
  for (int off = 128; off > 0; off >>= 1) {
    if (t < off) red[t] += red[t + off];
    __syncthreads();
  }
  const float mu = red[0] * (1.0f / H_DIM);
  __syncthreads();

  const float dx = v.x - mu, dy = v.y - mu, dz = v.z - mu, dw = v.w - mu;
  red[t] = dx * dx + dy * dy + dz * dz + dw * dw;
  __syncthreads();
#pragma unroll
  for (int off = 128; off > 0; off >>= 1) {
    if (t < off) red[t] += red[t + off];
    __syncthreads();
  }
  const float rstd = rsqrtf(red[0] * (1.0f / H_DIM) + 1e-5f);

  const float4 gv = ((const float4*)g)[t];
  const float4 bv = ((const float4*)bb)[t];
  v4bf o;
  o[0] = (bf16_t)(dx * rstd * gv.x + bv.x);
  o[1] = (bf16_t)(dy * rstd * gv.y + bv.y);
  o[2] = (bf16_t)(dz * rstd * gv.z + bv.z);
  o[3] = (bf16_t)(dw * rstd * gv.w + bv.w);
  *(v4bf*)(y + (size_t)blockIdx.x * H_DIM + (size_t)t * 4) = o;
}

// ---------------------------------------------------------------------------
// Utility kernels
// ---------------------------------------------------------------------------
__global__ __launch_bounds__(256) void cast_f32_bf16(
    const float* __restrict__ in, bf16_t* __restrict__ out, int n) {
  const int i = blockIdx.x * 256 + threadIdx.x;
  if (i < n) out[i] = (bf16_t)in[i];
}

__global__ __launch_bounds__(256) void fill_zero_f32(float* __restrict__ p, int n) {
  const int i = blockIdx.x * 256 + threadIdx.x;
  if (i < n) p[i] = 0.0f;
}

__global__ __launch_bounds__(256) void fill_zero_bf16(bf16_t* __restrict__ p, int n) {
  const int i = blockIdx.x * 256 + threadIdx.x;
  if (i < n) p[i] = (bf16_t)0.0f;
}

// ---------------------------------------------------------------------------
// Host orchestration
// ---------------------------------------------------------------------------
extern "C" void kernel_launch(void* const* d_in, const int* in_sizes, int n_in,
                              void* d_out, int out_size, void* d_ws, size_t ws_size,
                              hipStream_t stream) {
  (void)in_sizes; (void)n_in; (void)out_size; (void)ws_size;

  const float* x      = (const float*)d_in[0];
  const float* proj_W = (const float*)d_in[1];
  const float* proj_b = (const float*)d_in[2];
  const float* h_Wih  = (const float*)d_in[3];
  const float* h_Whh  = (const float*)d_in[4];
  const float* h_bih  = (const float*)d_in[5];
  const float* h_bhh  = (const float*)d_in[6];
  const float* ctrl_W = (const float*)d_in[7];
  const float* ctrl_b = (const float*)d_in[8];
  const float* h_ln_g = (const float*)d_in[9];
  const float* h_ln_b = (const float*)d_in[10];
  const float* l_Wih  = (const float*)d_in[11];
  const float* l_Whh  = (const float*)d_in[12];
  const float* l_bih  = (const float*)d_in[13];
  const float* l_bhh  = (const float*)d_in[14];
  const float* ref_W1 = (const float*)d_in[15];
  const float* ref_b1 = (const float*)d_in[16];
  const float* ref_W2 = (const float*)d_in[17];
  const float* ref_b2 = (const float*)d_in[18];
  const float* l_ln_g = (const float*)d_in[19];
  const float* l_ln_b = (const float*)d_in[20];
  const float* out_W1 = (const float*)d_in[21];
  const float* out_b1 = (const float*)d_in[22];
  const float* out_W2 = (const float*)d_in[23];
  const float* out_b2 = (const float*)d_in[24];
  float* out = (float*)d_out;

  // ---- workspace layout: bf16 (weights, x, activations), then f32 buffers --
  const size_t BH  = (size_t)B_DIM * H_DIM;
  const size_t XSZ = (size_t)B_DIM * S_DIM * D_DIM;

  bf16_t* wb  = (bf16_t*)d_ws;
  size_t  off = 0;
  auto bfseg = [&](size_t n) { bf16_t* p = wb + off; off += n; return p; };
  bf16_t* w_proj  = bfseg((size_t)H_DIM * D_DIM);
  bf16_t* w_hih   = bfseg((size_t)3 * H_DIM * H_DIM);
  bf16_t* w_hhh   = bfseg((size_t)3 * H_DIM * H_DIM);
  bf16_t* w_ctrl  = bfseg((size_t)H_DIM * H_DIM);
  bf16_t* w_lih   = bfseg((size_t)3 * H_DIM * H_DIM);
  bf16_t* w_lhh   = bfseg((size_t)3 * H_DIM * H_DIM);
  bf16_t* w_ref1  = bfseg((size_t)H_DIM * H_DIM);
  bf16_t* w_ref2  = bfseg((size_t)H_DIM * H_DIM);
  bf16_t* w_out1  = bfseg((size_t)H_DIM * H_DIM);
  bf16_t* w_out2  = bfseg((size_t)O_DIM * H_DIM);
  bf16_t* xbf     = bfseg(XSZ);
  bf16_t* xp_bf   = bfseg(BH);
  bf16_t* h_bf    = bfseg(BH);
  bf16_t* ctrl_bf = bfseg(BH);
  bf16_t* lh_bf   = bfseg(BH);
  bf16_t* r1_bf   = bfseg(BH);
  bf16_t* st_bf   = bfseg(BH);
  bf16_t* m_bf    = bfseg(BH);
  const size_t bfBytes = ((off * sizeof(bf16_t)) + 255) & ~(size_t)255;

  float* fp = (float*)((char*)d_ws + bfBytes);
  float* buf_gi   = fp; fp += 3 * BH;
  float* buf_gh   = fp; fp += 3 * BH;
  float* buf_h    = fp; fp += BH;
  float* buf_lh   = fp; fp += BH;
  float* buf_cpre = fp; fp += BH;
  float* buf_r2   = fp; fp += BH;

  // ---- helpers ----
  auto cast = [&](const float* src, bf16_t* dst, size_t n) {
    cast_f32_bf16<<<(unsigned)((n + 255) / 256), 256, 0, stream>>>(src, dst, (int)n);
  };
  auto gemm = [&](const bf16_t* A, long lda, const bf16_t* W, const float* bias,
                  void* C, int M, int N, int K, bool relu, bool outbf) {
    dim3 grid(N / BN, M / BM);
    if (relu && outbf)
      gemm_bf16_wmma<true, true><<<grid, 256, 0, stream>>>(A, lda, W, bias, C, M, N, K);
    else if (outbf)
      gemm_bf16_wmma<false, true><<<grid, 256, 0, stream>>>(A, lda, W, bias, C, M, N, K);
    else
      gemm_bf16_wmma<false, false><<<grid, 256, 0, stream>>>(A, lda, W, bias, C, M, N, K);
  };
  auto gru = [&](const float* gi, const float* gh, float* h, bf16_t* hb) {
    gru_fuse_kernel<<<(unsigned)((BH + 255) / 256), 256, 0, stream>>>(gi, gh, h, hb, (int)BH);
  };
  auto lnorm = [&](const float* src, const float* g, const float* b, bf16_t* dst) {
    layernorm_kernel<<<B_DIM, 256, 0, stream>>>(src, g, b, dst);
  };
  auto zero_f = [&](float* p, size_t n) {
    fill_zero_f32<<<(unsigned)((n + 255) / 256), 256, 0, stream>>>(p, (int)n);
  };
  auto zero_b = [&](bf16_t* p, size_t n) {
    fill_zero_bf16<<<(unsigned)((n + 255) / 256), 256, 0, stream>>>(p, (int)n);
  };

  // ---- 1. cast weights and x to bf16 ----
  cast(proj_W, w_proj, (size_t)H_DIM * D_DIM);
  cast(h_Wih,  w_hih,  (size_t)3 * H_DIM * H_DIM);
  cast(h_Whh,  w_hhh,  (size_t)3 * H_DIM * H_DIM);
  cast(ctrl_W, w_ctrl, (size_t)H_DIM * H_DIM);
  cast(l_Wih,  w_lih,  (size_t)3 * H_DIM * H_DIM);
  cast(l_Whh,  w_lhh,  (size_t)3 * H_DIM * H_DIM);
  cast(ref_W1, w_ref1, (size_t)H_DIM * H_DIM);
  cast(ref_W2, w_ref2, (size_t)H_DIM * H_DIM);
  cast(out_W1, w_out1, (size_t)H_DIM * H_DIM);
  cast(out_W2, w_out2, (size_t)O_DIM * H_DIM);
  cast(x, xbf, XSZ);

  // ---- 2. H-step 0: per-timestep projection + GRU scan over S ----
  zero_f(buf_h, BH);
  zero_b(h_bf, BH);
  for (int t = 0; t < S_DIM; ++t) {
    gemm(xbf + (long)t * D_DIM, (long)S_DIM * D_DIM, w_proj, proj_b,
         xp_bf, B_DIM, H_DIM, D_DIM, false, true);
    gemm(xp_bf, H_DIM, w_hih, h_bih, buf_gi, B_DIM, 3 * H_DIM, H_DIM, false, false);
    gemm(h_bf,  H_DIM, w_hhh, h_bhh, buf_gh, B_DIM, 3 * H_DIM, H_DIM, false, false);
    gru(buf_gi, buf_gh, buf_h, h_bf);
  }

  // ---- 3. hierarchical H/L steps ----
  for (int step = 0; step < NUM_STEPS; ++step) {
    if (step > 0) {
      gemm(st_bf, H_DIM, w_hih, h_bih, buf_gi, B_DIM, 3 * H_DIM, H_DIM, false, false);
      gemm(h_bf,  H_DIM, w_hhh, h_bhh, buf_gh, B_DIM, 3 * H_DIM, H_DIM, false, false);
      gru(buf_gi, buf_gh, buf_h, h_bf);
    }
    gemm(h_bf, H_DIM, w_ctrl, ctrl_b, buf_cpre, B_DIM, H_DIM, H_DIM, false, false);
    lnorm(buf_cpre, h_ln_g, h_ln_b, ctrl_bf);

    zero_f(buf_lh, BH);
    zero_b(lh_bf, BH);
    const bf16_t* state = ctrl_bf;
    for (int it = 0; it < L_ITERS; ++it) {
      gemm(state, H_DIM, w_lih, l_bih, buf_gi, B_DIM, 3 * H_DIM, H_DIM, false, false);
      gemm(lh_bf, H_DIM, w_lhh, l_bhh, buf_gh, B_DIM, 3 * H_DIM, H_DIM, false, false);
      gru(buf_gi, buf_gh, buf_lh, lh_bf);
      gemm(lh_bf, H_DIM, w_ref1, ref_b1, r1_bf, B_DIM, H_DIM, H_DIM, true, true);
      gemm(r1_bf, H_DIM, w_ref2, ref_b2, buf_r2, B_DIM, H_DIM, H_DIM, false, false);
      lnorm(buf_r2, l_ln_g, l_ln_b, st_bf);
      state = st_bf;
    }
  }

  // ---- 4. output head ----
  gemm(st_bf, H_DIM, w_out1, out_b1, m_bf, B_DIM, H_DIM, H_DIM, true, true);
  gemm(m_bf,  H_DIM, w_out2, out_b2, out,  B_DIM, O_DIM, H_DIM, false, false);
}
